// LinearMultiHeadedAttention_85933705658676
// MI455X (gfx1250) — matrix operand order repaired and checked by reading, not compiled, not run on hardware
//
#include <hip/hip_runtime.h>
#include <hip/hip_bf16.h>
#include <math.h>

typedef __bf16 bf16;
typedef __attribute__((ext_vector_type(16))) __bf16 v16bf;
typedef __attribute__((ext_vector_type(8)))  __bf16 v8bf;
typedef __attribute__((ext_vector_type(2)))  __bf16 v2bf;
typedef __attribute__((ext_vector_type(8)))  float  v8f;
typedef __attribute__((ext_vector_type(4)))  unsigned int v4u;
typedef __attribute__((ext_vector_type(8)))  int v8i;
typedef __attribute__((ext_vector_type(4)))  int v4i;

#define NB   16      // batch
#define SEQ  4096    // sequence length
#define DM   1024    // d_model
#define NH   16      // heads
#define DK   64      // head dim
#define SPLIT 8      // S-splits per (b,h)
#define CH   64      // rows per wave-chunk (4 M-tiles)
#define LDT  72      // stage-B LDS stride (halves): 144B rows, 16B aligned, bank-staggered
#define WSLAB_K   128                  // weight slab k-extent (elements)
#define WSLAB_LDS 136                  // slab row stride in halves (272B: TDM pad 16B per 256B row)
#define WSLAB_BYTES (DK * WSLAB_LDS * 2)   // 17408 B per slab buffer
#define STAGE_OFF   (2 * WSLAB_BYTES)      // 34816 B : start of per-wave stage-B tiles

#if __has_builtin(__builtin_amdgcn_tensor_load_to_lds)
#define USE_TDM 1
#else
#define USE_TDM 0
#endif

// ---------------- WMMA fragment helpers (contiguous k order: k = 16*(lane/16) + s for A and B) -----

// A-matrix 16x32 bf16 from row-major fp32 source (convert in-register).
__device__ __forceinline__ v16bf load_a_f32(const float* __restrict__ src, int ld,
                                            int m0, int k0, int lane) {
  int m  = m0 + (lane & 15);
  int hi = lane >> 4;
  const float* p  = src + (size_t)m * ld + k0 + hi * 16;
  const float4 x0 = ((const float4*)p)[0];
  const float4 x1 = ((const float4*)p)[1];
  const float4 y0 = ((const float4*)(p + 8))[0];
  const float4 y1 = ((const float4*)(p + 8))[1];
  v16bf a;
  a[0]=(bf16)x0.x; a[1]=(bf16)x0.y; a[2]=(bf16)x0.z; a[3]=(bf16)x0.w;
  a[4]=(bf16)x1.x; a[5]=(bf16)x1.y; a[6]=(bf16)x1.z; a[7]=(bf16)x1.w;
  a[8]=(bf16)y0.x; a[9]=(bf16)y0.y; a[10]=(bf16)y0.z; a[11]=(bf16)y0.w;
  a[12]=(bf16)y1.x; a[13]=(bf16)y1.y; a[14]=(bf16)y1.z; a[15]=(bf16)y1.w;
  return a;
}

__device__ __forceinline__ v16bf cat8(v8bf lo, v8bf hi) {
  v16bf r;
#pragma unroll
  for (int i = 0; i < 8; ++i) { r[i] = lo[i]; r[8 + i] = hi[i]; }
  return r;
}

// B-matrix 32x16 bf16 straight from global W stored [n][k] row-major (fallback path).
__device__ __forceinline__ v16bf load_b_w(const bf16* __restrict__ w, int ld,
                                          int k0, int n0, int lane) {
  int n = n0 + (lane & 15);
  int k = k0 + ((lane >> 4) << 4);
  return *(const v16bf*)(w + (size_t)n * ld + k);
}

// B-matrix 32x16 bf16 from a TDM-staged LDS weight slab (row stride WSLAB_LDS halves).
__device__ __forceinline__ v16bf lds_b_slab(const bf16* buf, int nt, int kl, int lane) {
  int n = nt * 16 + (lane & 15);
  const bf16* row = buf + n * WSLAB_LDS + kl + ((lane >> 4) << 4);
  v8bf lo = *(const v8bf*)(row);
  v8bf hb = *(const v8bf*)(row + 8);
  return cat8(lo, hb);
}

// A-frag (16x32) from stage tile stored transposed: T[d][s], stride LDT, window ks..ks+31
__device__ __forceinline__ v16bf lds_a(const bf16* t, int dt, int ks, int lane) {
  int md = dt * 16 + (lane & 15);
  int hi = lane >> 4;
  const bf16* row = t + md * LDT + ks;
  v8bf lo = *(const v8bf*)(row + hi * 16);
  v8bf hb = *(const v8bf*)(row + hi * 16 + 8);
  return cat8(lo, hb);
}

// B-frag (32x16) from stage tile stored transposed: T[m][s], stride LDT, window ks..ks+31
__device__ __forceinline__ v16bf lds_b(const bf16* t, int vt, int ks, int lane) {
  int nm = vt * 16 + (lane & 15);
  int hi = lane >> 4;
  const bf16* row = t + nm * LDT + ks + hi * 16;
  v8bf lo = *(const v8bf*)(row);
  v8bf hb = *(const v8bf*)(row + 8);
  return cat8(lo, hb);
}

__device__ __forceinline__ v8f wmma_bf16(v16bf a, v16bf b, v8f c) {
  return __builtin_amdgcn_wmma_f32_16x16x32_bf16(false, a, false, b, (short)0, c,
                                                 false, false);
}

#if USE_TDM
// Issue a TDM 2D tile load: 64 rows x 128 halves from W[h] (row stride DM halves)
// into LDS at lds_off, padded +16B per 256B row (-> 272B LDS row stride).
// D# layout per CDNA5 ISA. 6-arg builtin (this toolchain).
__device__ __forceinline__ void tdm_issue_wslab(const bf16* gtile, unsigned lds_off) {
  unsigned long long ga = (unsigned long long)(uintptr_t)gtile;
  v4u g0;
  g0[0] = 1u;                                        // count=1, user descriptor
  g0[1] = lds_off;                                   // LDS byte address
  g0[2] = (unsigned)(ga & 0xFFFFFFFFu);              // global_addr[31:0]
  g0[3] = (unsigned)((ga >> 32) & 0x1FFFFFFu) | (2u << 30);  // addr[56:32] | type=2
  v8i g1;
  g1[0] = (int)((1u << 16)      // data_size = 1 -> 2 bytes
              | (1u << 20)      // pad_enable
              | (5u << 22)      // pad_interval: 64 DWORDs (256B)
              | (3u << 25));    // pad_amount: 4 DWORDs (16B)
  g1[1] = (int)(((unsigned)DM & 0xFFFFu) << 16);     // tensor_dim0[15:0]=1024
  g1[2] = (int)(((unsigned)DK & 0xFFFFu) << 16);     // dim0 hi=0 | tensor_dim1[15:0]=64
  g1[3] = (int)(((unsigned)WSLAB_K) << 16);          // dim1 hi=0 | tile_dim0=128
  g1[4] = (int)DK;                                   // tile_dim1=64, tile_dim2=0
  g1[5] = (int)DM;                                   // tensor_dim0_stride[31:0]=1024
  g1[6] = 0;
  g1[7] = 0;                                         // tensor_dim1_stride=0 (2D)
  v4i gz4 = {0, 0, 0, 0};
  v8i gz8 = {0, 0, 0, 0, 0, 0, 0, 0};
  __builtin_amdgcn_tensor_load_to_lds(g0, g1, gz4, gz4, gz8, 0);
}
#endif

// ---------------- epilogue / stage-B helpers ----------------------------------------
__device__ __forceinline__ void epilogue_k(v8f (&acc)[4][4], bf16* Kt,
                                           const float* biask, float* ks, int lane) {
#pragma unroll
  for (int mt = 0; mt < 4; ++mt)
#pragma unroll
    for (int nt = 0; nt < 4; ++nt) {
      const int dk = nt * 16 + (lane & 15);
      bf16* dst = &Kt[dk * LDT + mt * 16 + ((lane >> 4) << 3)];
#pragma unroll
      for (int r = 0; r < 8; r += 2) {
        float x0 = acc[mt][nt][r]     + biask[nt];
        float x1 = acc[mt][nt][r + 1] + biask[nt];
        x0 = x0 > 0.0f ? x0 + 1.0f : __expf(x0);     // elu(x)+1
        x1 = x1 > 0.0f ? x1 + 1.0f : __expf(x1);
        ks[nt] += x0 + x1;
        v2bf p; p[0] = (bf16)x0; p[1] = (bf16)x1;
        *(v2bf*)(dst + r) = p;
      }
    }
}

__device__ __forceinline__ void epilogue_v(v8f (&acc)[4][4], bf16* Vt,
                                           const float* biasv, int lane) {
#pragma unroll
  for (int mt = 0; mt < 4; ++mt)
#pragma unroll
    for (int nt = 0; nt < 4; ++nt) {
      const int dm = nt * 16 + (lane & 15);
      bf16* dst = &Vt[dm * LDT + mt * 16 + ((lane >> 4) << 3)];
#pragma unroll
      for (int r = 0; r < 8; r += 2) {
        v2bf p;
        p[0] = (bf16)(acc[mt][nt][r]     + biasv[nt]);
        p[1] = (bf16)(acc[mt][nt][r + 1] + biasv[nt]);
        *(v2bf*)(dst + r) = p;
      }
    }
}

__device__ __forceinline__ void stage_b(const bf16* Kt, const bf16* Vt,
                                        v8f (&kv)[4][4], int lane) {
#pragma unroll
  for (int kss = 0; kss < 2; ++kss) {
    const int kst = kss * 32;
    v16bf bV[4];
#pragma unroll
    for (int vt = 0; vt < 4; ++vt) bV[vt] = lds_b(Vt, vt, kst, lane);
#pragma unroll
    for (int dt = 0; dt < 4; ++dt) {
      v16bf aK = lds_a(Kt, dt, kst, lane);
#pragma unroll
      for (int vt = 0; vt < 4; ++vt)
        kv[dt][vt] = wmma_bf16(aK, bV[vt], kv[dt][vt]);
    }
  }
}

// ---------------- prep: fp32 weights -> bf16, zero accumulators ----------------------
__global__ void prep_kernel(const float* __restrict__ Wk, const float* __restrict__ Wv,
                            bf16* __restrict__ wkb, bf16* __restrict__ wvb,
                            float* __restrict__ KV, float* __restrict__ ksum) {
  int i = blockIdx.x * blockDim.x + threadIdx.x;
  if (i < DM * DM) {
    wkb[i] = (bf16)Wk[i];
    wvb[i] = (bf16)Wv[i];
    KV[i]  = 0.0f;                                 // NB*NH*DK*DK == DM*DM
  }
  if (i < NB * NH * DK) ksum[i] = 0.0f;
}

// ---------------- fused projection + feature map + KV accumulation ------------------
__global__ void __launch_bounds__(128)
kv_kernel(const float* __restrict__ key, const float* __restrict__ value,
          const float* __restrict__ bk, const float* __restrict__ bv,
          const bf16* __restrict__ wkb, const bf16* __restrict__ wvb,
          float* __restrict__ KV, float* __restrict__ ksum) {
  extern __shared__ char dsm[];
  bf16* wslab = (bf16*)dsm;                                    // 2 slab buffers (TDM)
  const int split = blockIdx.x, h = blockIdx.y, b = blockIdx.z;
  const int lane = threadIdx.x & 31;
  const int w    = threadIdx.x >> 5;

  const float* keyb = key   + (size_t)b * SEQ * DM;
  const float* valb = value + (size_t)b * SEQ * DM;
  const bf16*  wk_h = wkb + (size_t)h * DK * DM;
  const bf16*  wv_h = wvb + (size_t)h * DK * DM;
  bf16* Kt = (bf16*)(dsm + STAGE_OFF) + (size_t)w * (2 * DK * LDT);
  bf16* Vt = Kt + DK * LDT;

  float biask[4], biasv[4];
#pragma unroll
  for (int nt = 0; nt < 4; ++nt) {
    biask[nt] = bk[h * DK + nt * 16 + (lane & 15)];
    biasv[nt] = bv[h * DK + nt * 16 + (lane & 15)];
  }

  v8f kv[4][4];
#pragma unroll
  for (int i = 0; i < 4; ++i)
#pragma unroll
    for (int j = 0; j < 4; ++j)
#pragma unroll
      for (int r = 0; r < 8; ++r) kv[i][j][r] = 0.0f;
  float ks[4] = {0.f, 0.f, 0.f, 0.f};

  const int rows = SEQ / SPLIT;                   // 512 rows per workgroup

#if USE_TDM
  // 4 waves cooperate on a 256-row macro-chunk; wave w owns rows [w*64, w*64+64).
  // Weight slabs (64n x 128k bf16) are TDM-DMA'd into double-buffered LDS by wave 0.
  for (int mc = 0; mc < rows / (4 * CH); ++mc) {  // 2 macro-chunks
    const int s0 = split * rows + mc * 4 * CH + w * CH;
    if (s0 + 4 * CH < SEQ) __builtin_prefetch(keyb + (size_t)(s0 + 4 * CH) * DM, 0, 0);

#pragma unroll
    for (int pass = 0; pass < 2; ++pass) {        // 0: K-proj(Wk), 1: V-proj(Wv)
      const float* src  = pass ? valb : keyb;
      const bf16*  wsrc = pass ? wv_h : wk_h;

      v8f acc[4][4];
#pragma unroll
      for (int i = 0; i < 4; ++i)
#pragma unroll
        for (int j = 0; j < 4; ++j)
#pragma unroll
          for (int r = 0; r < 8; ++r) acc[i][j][r] = 0.0f;

      if (w == 0) tdm_issue_wslab(wsrc, (unsigned)(uintptr_t)(void*)(wslab));
      for (int sl = 0; sl < DM / WSLAB_K; ++sl) { // 8 slabs
        if (w == 0) {
          if (sl < DM / WSLAB_K - 1) {
            tdm_issue_wslab(wsrc + (sl + 1) * WSLAB_K,
                            (unsigned)(uintptr_t)(void*)(wslab) +
                                (unsigned)(((sl + 1) & 1) * WSLAB_BYTES));
            __builtin_amdgcn_s_wait_tensorcnt(1); // slab sl complete, sl+1 in flight
          } else {
            __builtin_amdgcn_s_wait_tensorcnt(0);
          }
        }
        __syncthreads();                          // slab sl visible to all waves
        const bf16* buf = wslab + (sl & 1) * (DK * WSLAB_LDS);
#pragma unroll
        for (int ksub = 0; ksub < WSLAB_K / 32; ++ksub) {
          const int kk = sl * WSLAB_K + ksub * 32;
          v16bf a[4];
#pragma unroll
          for (int mt = 0; mt < 4; ++mt)
            a[mt] = load_a_f32(src, DM, s0 + mt * 16, kk, lane);
#pragma unroll
          for (int nt = 0; nt < 4; ++nt) {
            v16bf bw = lds_b_slab(buf, nt, ksub * 32, lane);
#pragma unroll
            for (int mt = 0; mt < 4; ++mt)
              acc[mt][nt] = wmma_bf16(a[mt], bw, acc[mt][nt]);
          }
        }
        __syncthreads();                          // all done reading before overwrite
      }
      if (pass == 0) epilogue_k(acc, Kt, biask, ks, lane);
      else           epilogue_v(acc, Vt, biasv, lane);
    }
    stage_b(Kt, Vt, kv, lane);                    // wave-private
  }
#else
  // Fallback: per-wave chunks, B-fragments straight from global bf16 weights.
  for (int c = w; c < rows / CH; c += 4) {
    const int s0 = split * rows + c * CH;
    if (s0 + 4 * CH < SEQ) __builtin_prefetch(keyb + (size_t)(s0 + 4 * CH) * DM, 0, 0);
#pragma unroll
    for (int pass = 0; pass < 2; ++pass) {
      const float* src  = pass ? valb : keyb;
      const bf16*  wsrc = pass ? wv_h : wk_h;
      v8f acc[4][4];
#pragma unroll
      for (int i = 0; i < 4; ++i)
#pragma unroll
        for (int j = 0; j < 4; ++j)
#pragma unroll
          for (int r = 0; r < 8; ++r) acc[i][j][r] = 0.0f;
      for (int kk = 0; kk < DM; kk += 32) {
        v16bf a[4];
#pragma unroll
        for (int mt = 0; mt < 4; ++mt)
          a[mt] = load_a_f32(src, DM, s0 + mt * 16, kk, lane);
#pragma unroll
        for (int nt = 0; nt < 4; ++nt) {
          v16bf bw = load_b_w(wsrc, DM, kk, nt * 16, lane);
#pragma unroll
          for (int mt = 0; mt < 4; ++mt)
            acc[mt][nt] = wmma_bf16(a[mt], bw, acc[mt][nt]);
        }
      }
      if (pass == 0) epilogue_k(acc, Kt, biask, ks, lane);
      else           epilogue_v(acc, Vt, biasv, lane);
    }
    stage_b(Kt, Vt, kv, lane);
  }
#endif

  // ---- reduce across SPLIT/waves with global float atomics ----
  float* kvh = KV + ((size_t)(b * NH + h)) * DK * DK;
#pragma unroll
  for (int dt = 0; dt < 4; ++dt)
#pragma unroll
    for (int vt = 0; vt < 4; ++vt)
#pragma unroll
      for (int r = 0; r < 8; ++r) {
        const int d = dt * 16 + ((lane >> 4) << 3) + r;
        const int m = vt * 16 + (lane & 15);
        atomicAdd(&kvh[d * DK + m], kv[dt][vt][r]);
      }
  float* ksh = ksum + (size_t)(b * NH + h) * DK;
#pragma unroll
  for (int nt = 0; nt < 4; ++nt)
    atomicAdd(&ksh[nt * 16 + (lane & 15)], ks[nt]);
}

// ---------------- finalize (tiny): q projection + feature map -----------------------
__global__ void qfeat_kernel(const float* __restrict__ q, const float* __restrict__ Wq,
                             const float* __restrict__ bq, float* __restrict__ qf) {
  int idx = blockIdx.x * blockDim.x + threadIdx.x;   // 16384
  int b = idx >> 10, j = idx & 1023;
  const float4* x = (const float4*)(q + (size_t)b * DM);
  const float4* w = (const float4*)(Wq + (size_t)j * DM);
  float s = 0.0f;
  for (int i = 0; i < DM / 4; ++i) {
    float4 a = x[i], ww = w[i];
    s += a.x * ww.x + a.y * ww.y + a.z * ww.z + a.w * ww.w;
  }
  s += bq[j];
  qf[idx] = s > 0.0f ? s + 1.0f : __expf(s);
}

__global__ void attnv_kernel(const float* __restrict__ qf, const float* __restrict__ KV,
                             const float* __restrict__ ksum, float* __restrict__ vf) {
  int bh = blockIdx.x;        // 256
  int m  = threadIdx.x;       // 64
  const float* qv  = qf   + (size_t)bh * DK;
  const float* ksh = ksum + (size_t)bh * DK;
  const float* kvp = KV   + (size_t)bh * DK * DK;
  float den = 1e-6f, num = 0.0f;
  for (int d = 0; d < DK; ++d) {
    float qd = qv[d];
    den += qd * ksh[d];
    num += qd * kvp[d * DK + m];
  }
  vf[bh * DK + m] = num / den;
}

__global__ void out_kernel(const float* __restrict__ vflat, const float* __restrict__ Wo,
                           const float* __restrict__ bo, float* __restrict__ out) {
  int idx = blockIdx.x * blockDim.x + threadIdx.x;   // 16384
  int b = idx >> 10, n = idx & 1023;
  const float4* x = (const float4*)(vflat + (size_t)b * DM);
  const float4* w = (const float4*)(Wo + (size_t)n * DM);
  float s = 0.0f;
  for (int i = 0; i < DM / 4; ++i) {
    float4 a = x[i], ww = w[i];
    s += a.x * ww.x + a.y * ww.y + a.z * ww.z + a.w * ww.w;
  }
  out[idx] = s + bo[n];
}

// ---------------- launcher ----------------------------------------------------------
extern "C" void kernel_launch(void* const* d_in, const int* in_sizes, int n_in,
                              void* d_out, int out_size, void* d_ws, size_t ws_size,
                              hipStream_t stream) {
  const float* query = (const float*)d_in[0];
  const float* key   = (const float*)d_in[1];
  const float* value = (const float*)d_in[2];
  const float* Wq = (const float*)d_in[3];
  const float* bq = (const float*)d_in[4];
  const float* Wk = (const float*)d_in[5];
  const float* bk = (const float*)d_in[6];
  const float* Wv = (const float*)d_in[7];
  const float* bv = (const float*)d_in[8];
  const float* Wo = (const float*)d_in[9];
  const float* bo = (const float*)d_in[10];
  float* out = (float*)d_out;

  char* ws = (char*)d_ws;
  bf16*  wkb   = (bf16*)(ws);                               // 2 MiB
  bf16*  wvb   = (bf16*)(ws + (2ull << 20));                // 2 MiB
  float* KV    = (float*)(ws + (4ull << 20));               // 4 MiB
  float* ksum  = (float*)(ws + (8ull << 20));               // 64 KiB
  float* qf    = (float*)(ws + (8ull << 20) + (64u << 10)); // 64 KiB
  float* vflat = (float*)(ws + (8ull << 20) + (128u << 10));// 64 KiB

  // slab buffers + per-wave stage tiles (reserve max regardless of device path)
  const size_t kv_lds = (size_t)STAGE_OFF + 4u * 2u * DK * LDT * sizeof(bf16); // 108544 B

  prep_kernel<<<dim3((DM * DM + 255) / 256), dim3(256), 0, stream>>>(Wk, Wv, wkb, wvb,
                                                                     KV, ksum);
  kv_kernel<<<dim3(SPLIT, NH, NB), dim3(128), kv_lds, stream>>>(key, value, bk, bv,
                                                                wkb, wvb, KV, ksum);
  qfeat_kernel<<<dim3(64), dim3(256), 0, stream>>>(query, Wq, bq, qf);
  attnv_kernel<<<dim3(NB * NH), dim3(64), 0, stream>>>(qf, KV, ksum, vflat);
  out_kernel<<<dim3(64), dim3(256), 0, stream>>>(vflat, Wo, bo, out);
}
